// AttenDense_76433238000392
// MI455X (gfx1250) — matrix operand order repaired and checked by reading, not compiled, not run on hardware
//
#include <hip/hip_runtime.h>
#include <math.h>

// ---------------------------------------------------------------------------
// AttenDense fused pipeline for MI455X (gfx1250, wave32, WMMA).
//   scores = xb @ Ab^T            (bf16 WMMA, f32 accum)
//   stats  = rowwise (max, 1/sum(exp))
//   xr     = x + softmax(scores) @ A     (softmax fused into A-tile load)
//   y      = leaky(xr @ W^T + b)
// GEMMs are "NT" (operands row-major, K contiguous) -> WMMA fragments are
// straight b128 LDS reads. Global->LDS uses gfx1250 async-to-LDS ops with a
// double-buffered LDS pipeline so tile traffic hides behind the WMMA pipe.
// ---------------------------------------------------------------------------

typedef __attribute__((ext_vector_type(16))) __bf16 v16bf;
typedef __attribute__((ext_vector_type(8)))  float  v8f;

#define THREADS 256
#define BM 128
#define BN 128
#define BK 32
#define BKP (BK + 8)   // padded LDS row: 80 bytes, 16B aligned, spreads banks

#if defined(__AMDGCN__) && __has_builtin(__builtin_amdgcn_global_load_async_to_lds_b128)
#define HAVE_ASYNC 1
#else
#define HAVE_ASYNC 0
#endif

union FragBF16 { v16bf v; uint4 u[2]; };

enum { MODE_SC = 0, MODE_AV = 1, MODE_FC = 2 };

#if HAVE_ASYNC
// The builtin wants: (v4i addrspace(1)*, v4i addrspace(3)*, imm offset, imm cpol)
typedef int v4i_vs __attribute__((vector_size(16)));
typedef __attribute__((address_space(1))) v4i_vs v4i_g;
typedef __attribute__((address_space(3))) v4i_vs v4i_l;

__device__ __forceinline__ v4i_g* as_global(const void* p) {
  return (v4i_g*)(p);
}
__device__ __forceinline__ v4i_l* as_lds(void* p) {
  return (v4i_l*)(p);
}
#endif

__device__ __forceinline__ void wait_async0() {
#if HAVE_ASYNC
#if __has_builtin(__builtin_amdgcn_s_wait_asynccnt)
  __builtin_amdgcn_s_wait_asynccnt(0);
#else
  asm volatile("s_wait_asynccnt 0" ::: "memory");
#endif
#endif
}

// ---- simple f32 -> bf16 convert -------------------------------------------
__global__ void k_cvt(const float* __restrict__ src, __bf16* __restrict__ dst, long n) {
  long i = (long)blockIdx.x * blockDim.x + threadIdx.x;
  long stride = (long)gridDim.x * blockDim.x;
  for (long j = i; j < n; j += stride) dst[j] = (__bf16)src[j];
}

// src [R][Cc] f32 row-major  ->  dst [Cc][R] bf16 (transposed copy)
__global__ void k_cvt_T(const float* __restrict__ src, __bf16* __restrict__ dst,
                        int R, int Cc) {
  long n = (long)R * Cc;
  long i = (long)blockIdx.x * blockDim.x + threadIdx.x;
  long stride = (long)gridDim.x * blockDim.x;
  for (long j = i; j < n; j += stride) {
    long r = j / Cc, c = j % Cc;
    dst[c * (long)R + r] = (__bf16)src[j];
  }
}

// ---- rowwise softmax statistics: stats[row] = (max, 1/sum(exp(s-max))) ----
__global__ void k_softmax_stats(const float* __restrict__ scores,
                                float2* __restrict__ stats, int C) {
  __shared__ float red[THREADS];
  const int tid = threadIdx.x;
  const long row = blockIdx.x;
  const float* s = scores + row * (long)C;

  float m = -INFINITY;
  for (int i = tid; i < C; i += THREADS) m = fmaxf(m, s[i]);
  red[tid] = m; __syncthreads();
  for (int off = THREADS / 2; off > 0; off >>= 1) {
    if (tid < off) red[tid] = fmaxf(red[tid], red[tid + off]);
    __syncthreads();
  }
  m = red[0]; __syncthreads();

  float sum = 0.0f;
  for (int i = tid; i < C; i += THREADS) sum += __expf(s[i] - m);
  red[tid] = sum; __syncthreads();
  for (int off = THREADS / 2; off > 0; off >>= 1) {
    if (tid < off) red[tid] += red[tid + off];
    __syncthreads();
  }
  if (tid == 0) stats[row] = make_float2(m, 1.0f / red[0]);
}

// ---- main WMMA GEMM: out[m,n] = sum_k Aop[m,k] * Bop[n,k]  (+ epilogue) ---
// 256 threads = 8 waves. Block tile 128x128, BK=32; 2x4 wave grid; each wave
// owns 64x32 = eight 16x16 f32 accumulators. LDS double-buffered; next tile
// streams in (async-to-LDS) while the current tile feeds the WMMA pipe.
template<int MODE>
__global__ void __launch_bounds__(THREADS)
k_gemm(const __bf16* __restrict__ Abf,      // SC/FC A operand [M][K] bf16
       const float*  __restrict__ Af32,     // AV: raw scores [M][K] f32
       const float2* __restrict__ stats,    // AV: per-row softmax stats
       const __bf16* __restrict__ Bbf,      // B operand [N][K] bf16
       const float*  __restrict__ resid,    // AV: residual x [M][N] f32
       const float*  __restrict__ bias,     // FC: bias [N]
       float*        __restrict__ outF,     // SC: scores   FC: d_out
       __bf16*       __restrict__ outB,     // AV: xr bf16
       int M, int N, int K) {
  __shared__ __bf16 At[2][BM][BKP];
  __shared__ __bf16 Bt[2][BN][BKP];

  const int tid  = threadIdx.x;
  const int wave = tid >> 5;
  const int lane = tid & 31;
  const int wm   = wave >> 2;         // 0..1 : 64 rows each
  const int wn   = wave & 3;          // 0..3 : 32 cols each
  const int m0   = blockIdx.y * BM;
  const int n0   = blockIdx.x * BN;
  const int lr   = lane & 15;
  const int lh   = lane >> 4;

  // cooperative-copy geometry: each thread owns rows r0 and r0+64 at col co
  const int r0 = tid >> 2;
  const int co = (tid & 3) * 8;

  v8f acc[4][2];
  const v8f vzero = {0.f, 0.f, 0.f, 0.f, 0.f, 0.f, 0.f, 0.f};
  #pragma unroll
  for (int i = 0; i < 4; ++i)
    #pragma unroll
    for (int j = 0; j < 2; ++j) acc[i][j] = vzero;

  // ---- staging state (fallback / AV paths) ----
  uint4  ra0, ra1, rb0, rb1;                   // bf16 reg staging
  float4 s0a, s0b, s1a, s1b;                   // AV: raw f32 scores
  float2 st0 = make_float2(0.f, 0.f), st1 = st0;
  if (MODE == MODE_AV) { st0 = stats[m0 + r0]; st1 = stats[m0 + r0 + 64]; }

  // Start fetch of tile at k0 into LDS buffer `b` (async) or regs (fallback).
  auto a_start = [&](int k0, int b) {
    if constexpr (MODE == MODE_AV) {
      const float* p0 = Af32 + (long)(m0 + r0) * K + k0 + co;
      const float* p1 = p0 + (long)64 * K;
      s0a = *(const float4*)p0; s0b = *(const float4*)(p0 + 4);
      s1a = *(const float4*)p1; s1b = *(const float4*)(p1 + 4);
      (void)b;
    } else {
#if HAVE_ASYNC
      const __bf16* g0 = Abf + (long)(m0 + r0) * K + k0 + co;
      __builtin_amdgcn_global_load_async_to_lds_b128(
          as_global(g0), as_lds(&At[b][r0][co]), 0, 0);
      __builtin_amdgcn_global_load_async_to_lds_b128(
          as_global(g0 + (long)64 * K), as_lds(&At[b][r0 + 64][co]), 0, 0);
#else
      ra0 = *(const uint4*)(Abf + (long)(m0 + r0) * K + k0 + co);
      ra1 = *(const uint4*)(Abf + (long)(m0 + r0 + 64) * K + k0 + co);
      (void)b;
#endif
    }
  };
  auto b_start = [&](int k0, int b) {
#if HAVE_ASYNC
    const __bf16* g0 = Bbf + (long)(n0 + r0) * K + k0 + co;
    __builtin_amdgcn_global_load_async_to_lds_b128(
        as_global(g0), as_lds(&Bt[b][r0][co]), 0, 0);
    __builtin_amdgcn_global_load_async_to_lds_b128(
        as_global(g0 + (long)64 * K), as_lds(&Bt[b][r0 + 64][co]), 0, 0);
#else
    rb0 = *(const uint4*)(Bbf + (long)(n0 + r0) * K + k0 + co);
    rb1 = *(const uint4*)(Bbf + (long)(n0 + r0 + 64) * K + k0 + co);
    (void)b;
#endif
  };
  // Commit staged registers into LDS buffer `b` (no-op on pure async paths).
  auto a_commit = [&](int b) {
    if constexpr (MODE == MODE_AV) {
      union { __bf16 t[8]; uint4 u; } pk;
      pk.t[0] = (__bf16)(__expf(s0a.x - st0.x) * st0.y);
      pk.t[1] = (__bf16)(__expf(s0a.y - st0.x) * st0.y);
      pk.t[2] = (__bf16)(__expf(s0a.z - st0.x) * st0.y);
      pk.t[3] = (__bf16)(__expf(s0a.w - st0.x) * st0.y);
      pk.t[4] = (__bf16)(__expf(s0b.x - st0.x) * st0.y);
      pk.t[5] = (__bf16)(__expf(s0b.y - st0.x) * st0.y);
      pk.t[6] = (__bf16)(__expf(s0b.z - st0.x) * st0.y);
      pk.t[7] = (__bf16)(__expf(s0b.w - st0.x) * st0.y);
      *(uint4*)&At[b][r0][co] = pk.u;
      pk.t[0] = (__bf16)(__expf(s1a.x - st1.x) * st1.y);
      pk.t[1] = (__bf16)(__expf(s1a.y - st1.x) * st1.y);
      pk.t[2] = (__bf16)(__expf(s1a.z - st1.x) * st1.y);
      pk.t[3] = (__bf16)(__expf(s1a.w - st1.x) * st1.y);
      pk.t[4] = (__bf16)(__expf(s1b.x - st1.x) * st1.y);
      pk.t[5] = (__bf16)(__expf(s1b.y - st1.x) * st1.y);
      pk.t[6] = (__bf16)(__expf(s1b.z - st1.x) * st1.y);
      pk.t[7] = (__bf16)(__expf(s1b.w - st1.x) * st1.y);
      *(uint4*)&At[b][r0 + 64][co] = pk.u;
    } else {
#if !HAVE_ASYNC
      *(uint4*)&At[b][r0][co]      = ra0;
      *(uint4*)&At[b][r0 + 64][co] = ra1;
#else
      (void)b;
#endif
    }
  };
  auto b_commit = [&](int b) {
#if !HAVE_ASYNC
    *(uint4*)&Bt[b][r0][co]      = rb0;
    *(uint4*)&Bt[b][r0 + 64][co] = rb1;
#else
    (void)b;
#endif
  };

  // Eight 16x16x32 WMMAs from LDS buffer `b`.
  auto compute = [&](int b) {
    // A fragment 16x32 bf16: lanes 0-15 -> K{0..7,16..23}, 16-31 -> K{8..15,24..31}
    FragBF16 af[4];
    #pragma unroll
    for (int i = 0; i < 4; ++i) {
      int row = wm * 64 + i * 16 + lr;
      int kg  = lh * 8;
      af[i].u[0] = *reinterpret_cast<const uint4*>(&At[b][row][kg]);
      af[i].u[1] = *reinterpret_cast<const uint4*>(&At[b][row][kg + 16]);
    }
    // B fragment 32x16 bf16: lane holds column (lane&15), lanes 0-15 -> K{0..7,16..23}, 16-31 -> K{8..15,24..31}
    FragBF16 bfr[2];
    #pragma unroll
    for (int j = 0; j < 2; ++j) {
      int col = wn * 32 + j * 16 + lr;
      int kg  = lh * 8;
      bfr[j].u[0] = *reinterpret_cast<const uint4*>(&Bt[b][col][kg]);
      bfr[j].u[1] = *reinterpret_cast<const uint4*>(&Bt[b][col][kg + 16]);
    }
    #pragma unroll
    for (int i = 0; i < 4; ++i)
      #pragma unroll
      for (int j = 0; j < 2; ++j)
        acc[i][j] = __builtin_amdgcn_wmma_f32_16x16x32_bf16(
            false, af[i].v, false, bfr[j].v, (short)0, acc[i][j], false, false);
  };

  // ---- software-pipelined main loop ----
  a_start(0, 0);
  b_start(0, 0);
  a_commit(0);
  b_commit(0);
  wait_async0();
  __syncthreads();

  int buf = 0;
  for (int k0 = 0; k0 < K; k0 += BK) {
    const int  nb       = buf ^ 1;
    const bool has_next = (k0 + BK) < K;
    if (has_next) {               // stream next tile while WMMAs run
      a_start(k0 + BK, nb);
      b_start(k0 + BK, nb);
    }
    if (k0 + 2 * BK < K) {        // gfx1250 global_prefetch_b8, 2 tiles ahead
      __builtin_prefetch(Bbf + (long)(n0 + r0) * K + (k0 + 2 * BK), 0, 1);
    }
    compute(buf);
    if (has_next) {
      a_commit(nb);
      b_commit(nb);
    }
    wait_async0();
    __syncthreads();
    buf = nb;
  }

  // Epilogue. C/D 16x16 f32 layout: VGPR r, lane L -> M = r + (L>>4)*8, N = L&15.
  const int mw = m0 + wm * 64;
  const int nw = n0 + wn * 32;
  #pragma unroll
  for (int i = 0; i < 4; ++i) {
    #pragma unroll
    for (int j = 0; j < 2; ++j) {
      const int col   = nw + j * 16 + lr;
      const int mbase = mw + i * 16 + lh * 8;
      const float bv  = (MODE == MODE_FC) ? bias[col] : 0.0f;
      #pragma unroll
      for (int r = 0; r < 8; ++r) {
        long idx = (long)(mbase + r) * N + col;
        float v = acc[i][j][r];
        if (MODE == MODE_SC) {
          outF[idx] = v;
        } else if (MODE == MODE_AV) {
          v += resid[idx];
          outB[idx] = (__bf16)v;
        } else {
          v += bv;
          outF[idx] = (v >= 0.0f) ? v : 0.01f * v;
        }
      }
    }
  }
}

// ---------------------------------------------------------------------------
extern "C" void kernel_launch(void* const* d_in, const int* in_sizes, int n_in,
                              void* d_out, int out_size, void* d_ws, size_t ws_size,
                              hipStream_t stream) {
  const float* x = (const float*)d_in[0];   // [8192, 1024]
  const float* A = (const float*)d_in[1];   // [4096, 1024]
  const float* W = (const float*)d_in[2];   // [4096, 1024]
  const float* b = (const float*)d_in[3];   // [4096]

  const int  D = 1024, C = 4096;
  const long Nrow = 8192;

  char* ws = (char*)d_ws;
  size_t off = 0;
  auto alloc = [&](size_t bytes) -> char* {
    char* p = ws + off;
    off += (bytes + 255) & ~(size_t)255;
    return p;
  };
  __bf16* xb     = (__bf16*)alloc(Nrow * D * 2);        // 16 MB
  __bf16* Ab     = (__bf16*)alloc((long)C * D * 2);     //  8 MB
  __bf16* AbT    = (__bf16*)alloc((long)C * D * 2);     //  8 MB  [D][C]
  __bf16* Wb     = (__bf16*)alloc((long)C * D * 2);     //  8 MB
  float*  scores = (float*)alloc(Nrow * C * 4);         // 128 MB
  float2* stats  = (float2*)alloc(Nrow * sizeof(float2));
  __bf16* xr     = (__bf16*)alloc(Nrow * D * 2);        // 16 MB

  // 1) precision drop to bf16 (GEMMs are matrix-op bound; bf16 doubles WMMA rate)
  k_cvt  <<<2048, THREADS, 0, stream>>>(x, xb, Nrow * D);
  k_cvt  <<<1024, THREADS, 0, stream>>>(A, Ab, (long)C * D);
  k_cvt  <<<1024, THREADS, 0, stream>>>(W, Wb, (long)C * D);
  k_cvt_T<<<1024, THREADS, 0, stream>>>(A, AbT, C, D);   // AbT[d][c] = A[c][d]

  // 2) scores = xb @ Ab^T      [8192 x 4096], K=1024
  dim3 gSC(C / BN, Nrow / BM);
  k_gemm<MODE_SC><<<gSC, THREADS, 0, stream>>>(
      xb, nullptr, nullptr, Ab, nullptr, nullptr, scores, nullptr,
      (int)Nrow, C, D);

  // 3) rowwise softmax stats
  k_softmax_stats<<<(int)Nrow, THREADS, 0, stream>>>(scores, stats, C);

  // 4) xr = x + softmax(scores) @ A     [8192 x 1024], K=4096
  dim3 gAV(D / BN, Nrow / BM);
  k_gemm<MODE_AV><<<gAV, THREADS, 0, stream>>>(
      nullptr, scores, stats, AbT, x, nullptr, nullptr, xr,
      (int)Nrow, D, C);

  // 5) y = leaky(xr @ Wb^T + b)         [8192 x 4096], K=1024
  dim3 gFC(C / BN, Nrow / BM);
  k_gemm<MODE_FC><<<gFC, THREADS, 0, stream>>>(
      xr, nullptr, nullptr, Wb, nullptr, b, (float*)d_out, nullptr,
      (int)Nrow, C, D);

  (void)in_sizes; (void)n_in; (void)out_size; (void)ws_size;
}
